// GraphSAGE_26560077759064
// MI455X (gfx1250) — matrix operand order repaired and checked by reading, not compiled, not run on hardware
//
#include <hip/hip_runtime.h>
#include <hip/hip_bf16.h>

// ---------------------------------------------------------------------------
// GraphSAGE (3-layer, mean agg) for MI455X / gfx1250.
//   agg:   fp32 global atomics (L2-resident: h/neigh working set << 192MB L2)
//   gemm:  v_wmma_f32_16x16x32_bf16, fp32 accumulate, 16x64 tile per wave,
//          bias+ReLU fused in epilogue
// ---------------------------------------------------------------------------

typedef __attribute__((ext_vector_type(16))) __bf16 v16bf;
typedef __attribute__((ext_vector_type(8)))  float  v8f;

union ABu { uint4 u[2]; v16bf v; };

__device__ __forceinline__ unsigned short f2bf(float f) {
  // round-to-nearest-even fp32 -> bf16
  unsigned int u = __float_as_uint(f);
  u += 0x7FFFu + ((u >> 16) & 1u);
  return (unsigned short)(u >> 16);
}

// -------------------------------- utility ----------------------------------

__global__ void zero_f32(float* __restrict__ p, long n) {
  long i = (long)blockIdx.x * blockDim.x + threadIdx.x;
  if (i < n) p[i] = 0.0f;
}

__global__ void deg_kernel(const int* __restrict__ dst, float* __restrict__ deg, int E) {
  int e = blockIdx.x * blockDim.x + threadIdx.x;
  if (e < E) atomicAdd(&deg[dst[e]], 1.0f);
}

// one thread per (edge, 4-float chunk): coalesced gather of h[src], scatter-add
__global__ void scatter_kernel(const float* __restrict__ h, const int* __restrict__ src,
                               const int* __restrict__ dst, float* __restrict__ neigh,
                               int E, int D) {
  int chunks = D >> 2;
  long gid = (long)blockIdx.x * blockDim.x + threadIdx.x;
  if (gid >= (long)E * chunks) return;
  int e = (int)(gid / chunks);
  int c = (int)(gid % chunks);
  int s = src[e];
  int d = dst[e];
  const float4 v = *(const float4*)(h + (size_t)s * D + (size_t)c * 4);
  float* o = neigh + (size_t)d * D + (size_t)c * 4;
  atomicAdd(o + 0, v.x);
  atomicAdd(o + 1, v.y);
  atomicAdd(o + 2, v.z);
  atomicAdd(o + 3, v.w);
}

__global__ void cvt_bf16(const float* __restrict__ in, unsigned short* __restrict__ out, long n) {
  long i = (long)blockIdx.x * blockDim.x + threadIdx.x;
  if (i < n) out[i] = f2bf(in[i]);
}

// neigh_bf16 = bf16(neigh * 1/max(deg,1))   (mean aggregation finalize)
__global__ void scale_cvt_bf16(const float* __restrict__ neigh, const float* __restrict__ deg,
                               unsigned short* __restrict__ out, int N, int D) {
  long i = (long)blockIdx.x * blockDim.x + threadIdx.x;
  if (i >= (long)N * D) return;
  int row = (int)(i / D);
  float inv = 1.0f / fmaxf(deg[row], 1.0f);
  out[i] = f2bf(neigh[i] * inv);
}

// Repack W [K x Nout] (row major, fp32) into per-lane B-fragment bf16 layout:
//   packed idx = ((kt*ntiles + nt)*32 + lane)*16 + t
//   col = nt*16 + (lane&15),  k = kt*32 + (t>=8 ? 16 : 0) + (lane>=16 ? 8 : 0) + (t&7)
__global__ void pack_w(const float* __restrict__ W, unsigned short* __restrict__ Wp,
                       int K, int Nout) {
  int idx = blockIdx.x * blockDim.x + threadIdx.x;
  if (idx >= K * Nout) return;
  int ntiles = Nout >> 4;
  int t    = idx & 15;
  int lane = (idx >> 4) & 31;
  int tile = idx >> 9;
  int nt = tile % ntiles;
  int kt = tile / ntiles;
  int col = nt * 16 + (lane & 15);
  int k   = kt * 32 + ((t >> 3) << 4) + ((lane >> 4) << 3) + (t & 7);
  Wp[idx] = f2bf(W[(size_t)k * Nout + col]);
}

// ------------------------------- WMMA GEMM ---------------------------------

// A fragment (16x32 bf16, row-major source, stride Kdim), per ISA layout:
//   lanes 0-15:  row = lane,    K = {kb..kb+7, kb+16..kb+23}, kb = kt*32
//   lanes 16-31: row = lane-16, K shifted by +8
__device__ __forceinline__ v16bf load_a(const unsigned short* __restrict__ base,
                                        int Kdim, int mtile, int kt, int lane) {
  int row = mtile * 16 + (lane & 15);
  int kb  = kt * 32 + ((lane >> 4) << 3);          // +0 or +8
  const unsigned short* p = base + (size_t)row * Kdim + kb;
  ABu r;
  r.u[0] = *(const uint4*)(p);                     // K kb .. kb+7
  r.u[1] = *(const uint4*)(p + 16);                // K kb+16 .. kb+23
  return r.v;
}

// B fragment from pre-packed weights: 16 contiguous halves per lane (32B)
__device__ __forceinline__ v16bf load_b(const unsigned short* __restrict__ Wp,
                                        int ntiles, int kt, int nt, int lane) {
  const unsigned short* p = Wp + (((size_t)(kt * ntiles + nt) * 32) + lane) * 16;
  ABu r;
  r.u[0] = *(const uint4*)(p);
  r.u[1] = *(const uint4*)(p + 8);
  return r.v;
}

// out[16 x 64 tile] = Abf @ WpS + Gbf @ WpN + bias  (optionally ReLU)
// one wave per 16x64 tile (4 ntiles, 4 f32 accumulators); per k-step the two
// A fragments are reused by 8 WMMAs -> 2.5 b128 loads per WMMA.
// 2 waves (64 threads) per block.
template <bool RELU>
__global__ __launch_bounds__(64)
void sage_gemm(const unsigned short* __restrict__ Abf,   // self features, bf16 [M x K]
               const unsigned short* __restrict__ Gbf,   // mean-agg features, bf16 [M x K]
               const unsigned short* __restrict__ WpS,   // packed W_self
               const unsigned short* __restrict__ WpN,   // packed W_neigh
               const float* __restrict__ bias,
               float* __restrict__ out,                  // [M x Nout] fp32
               int Kdim, int Nout) {
  int lane   = threadIdx.x & 31;
  int wave   = threadIdx.x >> 5;
  int mtile  = blockIdx.x;
  int nt0    = (blockIdx.y * 2 + wave) * 4;   // first of 4 consecutive ntiles
  int ntiles = Nout >> 4;

  v8f c[4];
#pragma unroll
  for (int t = 0; t < 4; ++t) c[t] = (v8f){0.f, 0.f, 0.f, 0.f, 0.f, 0.f, 0.f, 0.f};

  int ksteps = Kdim >> 5;
  for (int kt = 0; kt < ksteps; ++kt) {
    v16bf as = load_a(Abf, Kdim, mtile, kt, lane);
    v16bf an = load_a(Gbf, Kdim, mtile, kt, lane);
#pragma unroll
    for (int t = 0; t < 4; ++t) {
      v16bf bs = load_b(WpS, ntiles, kt, nt0 + t, lane);
      c[t] = __builtin_amdgcn_wmma_f32_16x16x32_bf16(false, as, false, bs,
                                                     (short)0, c[t], false, false);
      v16bf bn = load_b(WpN, ntiles, kt, nt0 + t, lane);
      c[t] = __builtin_amdgcn_wmma_f32_16x16x32_bf16(false, an, false, bn,
                                                     (short)0, c[t], false, false);
    }
  }

  // C/D layout: VGPR j -> row 16*mtile + j + (lane>=16 ? 8 : 0), col = 16*nt + (lane&15)
  int rbase = mtile * 16 + ((lane >> 4) << 3);
#pragma unroll
  for (int t = 0; t < 4; ++t) {
    int col = (nt0 + t) * 16 + (lane & 15);
    float bv = bias[col];
#pragma unroll
    for (int j = 0; j < 8; ++j) {
      float r = c[t][j] + bv;
      if (RELU) r = fmaxf(r, 0.0f);
      out[(size_t)(rbase + j) * Nout + col] = r;
    }
  }
}

// -------------------------------- launch -----------------------------------

extern "C" void kernel_launch(void* const* d_in, const int* in_sizes, int n_in,
                              void* d_out, int out_size, void* d_ws, size_t ws_size,
                              hipStream_t stream) {
  (void)in_sizes; (void)n_in; (void)out_size; (void)ws_size;

  const float* x   = (const float*)d_in[0];
  const int*   src = (const int*)d_in[1];
  const int*   dst = (const int*)d_in[2];
  const float* W_in[6] = {
    (const float*)d_in[3], (const float*)d_in[4],   // Ws0, Wn0
    (const float*)d_in[6], (const float*)d_in[7],   // Ws1, Wn1
    (const float*)d_in[9], (const float*)d_in[10],  // Ws2, Wn2
  };
  const float* b0 = (const float*)d_in[5];
  const float* b1 = (const float*)d_in[8];
  const float* b2 = (const float*)d_in[11];

  const int N = 50000;   // == 3125 * 16, no M tail
  const int E = 800000;

  // ---- workspace carve (256B aligned regions) ----
  char* w = (char*)d_ws;
  auto alloc = [&](size_t bytes) -> char* {
    char* p = w;
    w += (bytes + 255) & ~(size_t)255;
    return p;
  };
  float* deg            = (float*)alloc((size_t)N * 4);
  float* h1             = (float*)alloc((size_t)N * 256 * 4);
  float* h2             = (float*)alloc((size_t)N * 256 * 4);
  float* neigh          = (float*)alloc((size_t)N * 256 * 4);
  unsigned short* hbf   = (unsigned short*)alloc((size_t)N * 256 * 2);
  unsigned short* nbf   = (unsigned short*)alloc((size_t)N * 256 * 2);
  const int wk[3]  = {128, 256, 256};   // K per layer
  const int wn[3]  = {256, 256, 128};   // Nout per layer
  unsigned short* Wp[6];
  for (int i = 0; i < 3; ++i) {
    Wp[2 * i]     = (unsigned short*)alloc((size_t)wk[i] * wn[i] * 2);
    Wp[2 * i + 1] = (unsigned short*)alloc((size_t)wk[i] * wn[i] * 2);
  }

  // ---- pack weights into B-fragment layout (once per launch) ----
  for (int i = 0; i < 6; ++i) {
    int K = wk[i / 2], Nout = wn[i / 2];
    int total = K * Nout;
    pack_w<<<(total + 255) / 256, 256, 0, stream>>>(W_in[i], Wp[i], K, Nout);
  }

  // ---- degrees (shared by all layers) ----
  zero_f32<<<(N + 255) / 256, 256, 0, stream>>>(deg, N);
  deg_kernel<<<(E + 255) / 256, 256, 0, stream>>>(dst, deg, E);

  // ---- one SAGE layer ----
  auto run_layer = [&](const float* hin, int Din, int Dout,
                       const unsigned short* WpS, const unsigned short* WpN,
                       const float* bias, float* hout, bool relu) {
    long nel = (long)N * Din;
    int  eg  = (int)((nel + 255) / 256);
    zero_f32<<<eg, 256, 0, stream>>>(neigh, nel);
    long sthreads = (long)E * (Din >> 2);
    scatter_kernel<<<(int)((sthreads + 255) / 256), 256, 0, stream>>>(hin, src, dst,
                                                                      neigh, E, Din);
    cvt_bf16<<<eg, 256, 0, stream>>>(hin, hbf, nel);
    scale_cvt_bf16<<<eg, 256, 0, stream>>>(neigh, deg, nbf, N, Din);
    // one wave per 16x64 tile; 2 waves per block
    dim3 grid(N / 16, Dout / 128);
    if (relu)
      sage_gemm<true><<<grid, 64, 0, stream>>>(hbf, nbf, WpS, WpN, bias, hout, Din, Dout);
    else
      sage_gemm<false><<<grid, 64, 0, stream>>>(hbf, nbf, WpS, WpN, bias, hout, Din, Dout);
  };

  run_layer(x,  128, 256, Wp[0], Wp[1], b0, h1, true);
  run_layer(h1, 256, 256, Wp[2], Wp[3], b1, h2, true);
  run_layer(h2, 256, 128, Wp[4], Wp[5], b2, (float*)d_out, false);
}
